// LSTM_60181081751925
// MI455X (gfx1250) — matrix operand order repaired and checked by reading, not compiled, not run on hardware
//
#include <hip/hip_runtime.h>

// ---------------------------------------------------------------------------
// LSTM (IN=14, H=28, OUT=2, B=4096, T=512) for MI455X / gfx1250.
//
// Strategy:
//  * Do NOT materialize xg [T,B,4H] (940 MB -> ~80us of HBM at 23.3 TB/s).
//    Fold the input projection into the recurrent GEMM:
//       gates[16x112] = [h | x_t](16x42, pad 64) @ Wcat(64x112) + (b_ih+b_hh)
//    HBM traffic collapses to one pass over x (117 MB, ~5us).
//  * One wave32 owns a 16-row batch tile for all 512 steps. 256 independent
//    waves, one per block -> one per WGP: the scan is latency-bound, so each
//    recurrence wave gets a full SIMD's issue slots and private LDS bandwidth.
//  * Matmul via v_wmma_f32_16x16x32_f16: 7 N-tiles x 2 K-chunks = 14 WMMA/step,
//    f16 operands, fp32 accumulate. Cell math (sigmoid/tanh, c-state) is fp32
//    in registers.
//  * Weights live in LDS as f16, column-major over K, matching the 32x16
//    B-matrix VGPR layout (lane = N column, two 8-K runs per half-wave
//    -> two ds_load_b128 per chunk).
//  * h/x live in a per-wave LDS A-tile, row-major [16][64] f16, matching the
//    16-bit A layout (two ds_load_b128 per K-chunk per lane).
//  * x_{t+1} is prefetched into registers with global loads while step t's
//    WMMAs run.
// ---------------------------------------------------------------------------

typedef __attribute__((ext_vector_type(16))) _Float16 v16h;
typedef __attribute__((ext_vector_type(8)))  _Float16 v8h;
typedef __attribute__((ext_vector_type(8)))  float    v8f;

#define IN_F    14
#define HID     28
#define G4      112          // 4*HID
#define KPAD    64           // 28 (h) + 14 (x) + 22 zero pad
#define T_STEPS 512
#define BATCH   4096
#define WAVES   1
#define BLOCK   (WAVES * 32)

union H2U { _Float16 h[2]; unsigned u; };

__device__ __forceinline__ float fsigmoid(float v) {
    return __builtin_amdgcn_rcpf(1.0f + __expf(-v));
}
__device__ __forceinline__ float ftanh(float v) {
    return 2.0f * __builtin_amdgcn_rcpf(1.0f + __expf(-2.0f * v)) - 1.0f;
}

__global__ __launch_bounds__(BLOCK) void lstm_scan_kernel(
    const float* __restrict__ x,      // [B,T,IN]
    const float* __restrict__ W_ih,   // [112,14]
    const float* __restrict__ W_hh,   // [112,28]
    const float* __restrict__ b_ih,   // [112]
    const float* __restrict__ b_hh,   // [112]
    const float* __restrict__ W_out,  // [2,28]
    const float* __restrict__ b_out,  // [2]
    float* __restrict__ out)          // [B,2]
{
    __shared__ _Float16 wCol[G4 * KPAD];        // wCol[n*64 + k] = Wcat[k][n]
    __shared__ _Float16 aT[WAVES][16 * KPAD];   // per-wave A tile [16][64] f16
    __shared__ float    gT[WAVES][16 * G4];     // per-wave gate scratch [16][112]

    const int tid  = threadIdx.x;
    const int wave = tid >> 5;
    const int lane = tid & 31;
    const int half = lane >> 4;       // 0: lanes 0-15, 1: lanes 16-31
    const int lcol = lane & 15;

    // ---- combined weight matrix -> LDS (f16, K-major per column n) ----
    // Wcat[k][n]: k<28 -> W_hh[n][k]; 28<=k<42 -> W_ih[n][k-28]; else 0
    for (int idx = tid; idx < G4 * KPAD; idx += BLOCK) {
        const int n = idx >> 6, k = idx & 63;
        float v = 0.0f;
        if (k < HID)            v = W_hh[n * HID + k];
        else if (k < HID+IN_F)  v = W_ih[n * IN_F + (k - HID)];
        wCol[idx] = (_Float16)v;
    }
    __syncthreads();

    _Float16* aTile = &aT[wave][0];
    float*    gTile = &gT[wave][0];

    // zero A tile: h_{-1}=0, zero-pad tail (x region overwritten below)
    for (int i = lane; i < 16 * KPAD; i += 32) aTile[i] = (_Float16)0.0f;

    const int r0 = (blockIdx.x * WAVES + wave) * 16;   // batch-tile base row

    // per-lane gate bias for each N-tile (C-matrix init; column = lcol)
    float biasv[7];
    #pragma unroll
    for (int nt = 0; nt < 7; ++nt) {
        const int n = nt * 16 + lcol;
        biasv[nt] = b_ih[n] + b_hh[n];
    }

    // cell/hidden state, fp32: lane owns (row m=lcol, j = half*14 + e)
    const int mrow = lcol;
    const int jb   = half * 14;
    float creg[14], hreg[14];
    #pragma unroll
    for (int e = 0; e < 14; ++e) { creg[e] = 0.0f; hreg[e] = 0.0f; }

    // x staging: lanes 0..15 own batch row m16; 7 float2 (14 floats) per step
    const int m16 = lcol;
    const float2* xrow =
        (const float2*)(x + (size_t)(r0 + m16) * T_STEPS * IN_F);

    // prologue: x_0 -> A tile (k = 28..41)
    if (lane < 16) {
        #pragma unroll
        for (int i = 0; i < 7; ++i) {
            const float2 v = xrow[i];
            H2U p; p.h[0] = (_Float16)v.x; p.h[1] = (_Float16)v.y;
            *reinterpret_cast<unsigned*>(&aTile[m16 * KPAD + HID + 2*i]) = p.u;
        }
    }

    for (int t = 0; t < T_STEPS; ++t) {
        // ---- prefetch x_{t+1} into registers (hides under WMMAs) ----
        const int tn = (t + 1 < T_STEPS) ? (t + 1) : t;
        float2 xnext[7];
        if (lane < 16) {
            #pragma unroll
            for (int i = 0; i < 7; ++i)
                xnext[i] = xrow[(size_t)tn * 7 + i];
        }

        // ---- load A = [h_{t-1} | x_t | 0] per the 16-bit A VGPR layout ----
        // elem e<8 : k = kb + half*8 + e ; elem e>=8 : k = kb + 16 + half*8 + (e-8)
        v16h a0, a1;
        {
            const _Float16* ab = &aTile[(lane & 15) * KPAD];
            union { v16h v; v8h h[2]; } ua;
            ua.h[0] = *(const v8h*)(ab +      half * 8);
            ua.h[1] = *(const v8h*)(ab + 16 + half * 8);
            a0 = ua.v;
            ua.h[0] = *(const v8h*)(ab + 32 + half * 8);
            ua.h[1] = *(const v8h*)(ab + 48 + half * 8);
            a1 = ua.v;
        }

        // ---- gates = A @ Wcat + bias : 7 N-tiles x 2 K-chunks of WMMA ----
        v8f acc[7];
        #pragma unroll
        for (int nt = 0; nt < 7; ++nt) {
            v8f c;
            #pragma unroll
            for (int e = 0; e < 8; ++e) c[e] = biasv[nt];
            // B 32x16 layout: lane = column n, elem e<8 -> k = kb + half*8 + e ; e>=8 -> k = kb + 16 + half*8 + (e-8)
            const _Float16* wb = &wCol[(nt * 16 + lcol) * KPAD];
            union { v16h v; v8h h[2]; } ub;
            ub.h[0] = *(const v8h*)(wb + half * 8);
            ub.h[1] = *(const v8h*)(wb + 16 + half * 8);
            const v16h b0 = ub.v;
            ub.h[0] = *(const v8h*)(wb + 32 + half * 8);
            ub.h[1] = *(const v8h*)(wb + 48 + half * 8);
            const v16h b1 = ub.v;
            c = __builtin_amdgcn_wmma_f32_16x16x32_f16(
                    false, a0, false, b0, (short)0, c, false, false);
            c = __builtin_amdgcn_wmma_f32_16x16x32_f16(
                    false, a1, false, b1, (short)0, c, false, false);
            acc[nt] = c;
        }

        // ---- scatter D tiles to gate scratch (C/D layout: m = v + 8*half) ----
        #pragma unroll
        for (int nt = 0; nt < 7; ++nt) {
            const int n = nt * 16 + lcol;
            #pragma unroll
            for (int v = 0; v < 8; ++v)
                gTile[(v + 8 * half) * G4 + n] = acc[nt][v];
        }

        // ---- fp32 LSTM cell update: lane handles (mrow, j=jb..jb+13) ----
        #pragma unroll
        for (int e = 0; e < 14; ++e) {
            const int j = jb + e;
            const float gi = gTile[mrow * G4 +       j];
            const float gf = gTile[mrow * G4 +  28 + j];
            const float gg = gTile[mrow * G4 +  56 + j];
            const float go = gTile[mrow * G4 +  84 + j];
            const float cn = fsigmoid(gf) * creg[e] + fsigmoid(gi) * ftanh(gg);
            creg[e] = cn;
            hreg[e] = fsigmoid(go) * ftanh(cn);
        }

        // ---- h_t -> A tile (k = 0..27) as packed f16 pairs ----
        #pragma unroll
        for (int e = 0; e < 14; e += 2) {
            H2U p; p.h[0] = (_Float16)hreg[e]; p.h[1] = (_Float16)hreg[e + 1];
            *reinterpret_cast<unsigned*>(&aTile[mrow * KPAD + jb + e]) = p.u;
        }

        // ---- commit x_{t+1} -> A tile (k = 28..41) ----
        if (lane < 16) {
            #pragma unroll
            for (int i = 0; i < 7; ++i) {
                H2U p;
                p.h[0] = (_Float16)xnext[i].x;
                p.h[1] = (_Float16)xnext[i].y;
                *reinterpret_cast<unsigned*>(&aTile[m16 * KPAD + HID + 2*i]) = p.u;
            }
        }
    }

    // ---- out = h_T @ W_out^T + b_out  (fp32 h kept in registers) ----
    float s0 = 0.0f, s1 = 0.0f;
    #pragma unroll
    for (int e = 0; e < 14; ++e) {
        const int j = jb + e;
        s0 += hreg[e] * W_out[0 * HID + j];
        s1 += hreg[e] * W_out[1 * HID + j];
    }
    s0 += __shfl_xor(s0, 16, 32);   // combine the two half-wave j-ranges
    s1 += __shfl_xor(s1, 16, 32);
    if (half == 0) {
        float2 o;
        o.x = s0 + b_out[0];
        o.y = s1 + b_out[1];
        *(float2*)&out[(size_t)(r0 + mrow) * 2] = o;
    }
}

extern "C" void kernel_launch(void* const* d_in, const int* in_sizes, int n_in,
                              void* d_out, int out_size, void* d_ws, size_t ws_size,
                              hipStream_t stream) {
    (void)in_sizes; (void)n_in; (void)out_size; (void)d_ws; (void)ws_size;
    const float* x     = (const float*)d_in[0];
    const float* W_ih  = (const float*)d_in[1];
    const float* W_hh  = (const float*)d_in[2];
    const float* b_ih  = (const float*)d_in[3];
    const float* b_hh  = (const float*)d_in[4];
    const float* W_out = (const float*)d_in[5];
    const float* b_out = (const float*)d_in[6];
    float* out = (float*)d_out;

    dim3 grid(BATCH / (16 * WAVES));   // 256 blocks, 1 wave each -> 1 per WGP
    lstm_scan_kernel<<<grid, BLOCK, 0, stream>>>(
        x, W_ih, W_hh, b_ih, b_hh, W_out, b_out, out);
}
